// OneDirectionalSRUModel_74749610819841
// MI455X (gfx1250) — matrix operand order repaired and checked by reading, not compiled, not run on hardware
//
#include <hip/hip_runtime.h>
#include <hip/hip_bf16.h>

#define T_LEN 2048
#define B_SZ  8
#define D_SZ  512
#define N3    1536                  // 3*D
#define M_SZ  (T_LEN * B_SZ)        // 16384
#define L_SZ  2

#define BM 128
#define BN 128
#define BK 32
#define AK 40                       // padded LDS row length (80 B, 16B aligned)

typedef __attribute__((ext_vector_type(16))) __bf16 bf16x16;
typedef __attribute__((ext_vector_type(8)))  __bf16 bf16x8;
typedef __attribute__((ext_vector_type(4)))  __bf16 bf16x4;
typedef __attribute__((ext_vector_type(8)))  float  f32x8;

// -------------------- embedding gather: (B,T)->(T,B,D) --------------------
__global__ void sru_gather(const int* __restrict__ ids,
                           const float* __restrict__ emb,
                           float* __restrict__ x) {
  int linear = blockIdx.x * blockDim.x + threadIdx.x;   // over M*D/4
  int d4 = linear & 127;                                // D/4 = 128
  int tb = linear >> 7;                                 // t*B + b
  int b  = tb & (B_SZ - 1);
  int t  = tb >> 3;
  int id = ids[b * T_LEN + t];
  const float4 v = *(const float4*)(emb + (size_t)id * D_SZ + d4 * 4);
  *(float4*)(x + (size_t)tb * D_SZ + d4 * 4) = v;
}

// -------------------- GEMM: U(M,1536) = X(M,512) * W(512,1536) -------------
// fp32 emulated via bf16 hi/lo split: A*B ~= Ah*Bh + Ah*Bl + Al*Bh
__global__ void __launch_bounds__(256)
sru_gemm(const float* __restrict__ X, const float* __restrict__ W,
         float* __restrict__ U) {
  // A tiles row-major [m][k]; B tiles TRANSPOSED [n][k] so fragments are
  // contiguous 16B runs -> ds_load_b128.
  __shared__ alignas(16) __bf16 Ah[BM][AK];
  __shared__ alignas(16) __bf16 Al[BM][AK];
  __shared__ alignas(16) __bf16 Bth[BN][AK];
  __shared__ alignas(16) __bf16 Btl[BN][AK];

  const int tid  = threadIdx.x;
  const int wave = tid >> 5;
  const int lane = tid & 31;
  const int m0 = blockIdx.y * BM;
  const int n0 = blockIdx.x * BN;

  f32x8 acc[8] = {};

  const int mrow    = wave * 16 + (lane & 15);
  const int khalf   = lane >> 4;       // 0: lanes 0-15, 1: lanes 16-31
  const int ncol_lo = lane & 15;

  for (int k0 = 0; k0 < D_SZ; k0 += BK) {
    // cooperative load A tile (BM x BK): vectorized b64 LDS stores
#pragma unroll
    for (int i = 0; i < 4; ++i) {
      int linear = tid + i * 256;          // 0..1023 float4's
      int row = linear >> 3;               // BK/4 = 8 quads per row
      int kq  = linear & 7;
      const float4 v = *(const float4*)(X + (size_t)(m0 + row) * D_SZ + k0 + kq * 4);
      float vv[4] = {v.x, v.y, v.z, v.w};
      bf16x4 h4, l4;
#pragma unroll
      for (int j = 0; j < 4; ++j) {
        float f = vv[j];
        __bf16 hi = (__bf16)f;
        h4[j] = hi;
        l4[j] = (__bf16)(f - (float)hi);
      }
      *(bf16x4*)&Ah[row][kq * 4] = h4;
      *(bf16x4*)&Al[row][kq * 4] = l4;
    }
    // cooperative load B tile (BK x BN) from W (row stride N3), store transposed
#pragma unroll
    for (int i = 0; i < 4; ++i) {
      int linear = tid + i * 256;
      int krow = linear >> 5;              // BN/4 = 32 quads per row
      int nq   = linear & 31;
      const float4 v = *(const float4*)(W + (size_t)(k0 + krow) * N3 + n0 + nq * 4);
      float vv[4] = {v.x, v.y, v.z, v.w};
#pragma unroll
      for (int j = 0; j < 4; ++j) {
        float f = vv[j];
        __bf16 hi = (__bf16)f;
        Bth[nq * 4 + j][krow] = hi;
        Btl[nq * 4 + j][krow] = (__bf16)(f - (float)hi);
      }
    }
    __syncthreads();

    // A fragment: element j -> K = khalf*8 + j (j<8), 16 + khalf*8 + (j-8) (j>=8)
    // = two contiguous 16B runs -> 2x ds_load_b128 each
    bf16x16 a_hi, a_lo;
    {
      const __bf16* arh = &Ah[mrow][0];
      const __bf16* arl = &Al[mrow][0];
      bf16x8 h0 = *(const bf16x8*)(arh + khalf * 8);
      bf16x8 h1 = *(const bf16x8*)(arh + 16 + khalf * 8);
      bf16x8 l0 = *(const bf16x8*)(arl + khalf * 8);
      bf16x8 l1 = *(const bf16x8*)(arl + 16 + khalf * 8);
      a_hi = __builtin_shufflevector(h0, h1, 0,1,2,3,4,5,6,7,8,9,10,11,12,13,14,15);
      a_lo = __builtin_shufflevector(l0, l1, 0,1,2,3,4,5,6,7,8,9,10,11,12,13,14,15);
    }

#pragma unroll
    for (int s = 0; s < 8; ++s) {
      // B fragment: element j -> K = khalf*8 + j (j<8), 16 + khalf*8 + (j-8) (j>=8)
      // for column nc -> 2x ds_load_b128 each (from transposed tile)
      const int nc = s * 16 + ncol_lo;
      const __bf16* brh = &Bth[nc][khalf * 8];
      const __bf16* brl = &Btl[nc][khalf * 8];
      bf16x8 h0 = *(const bf16x8*)(brh);
      bf16x8 h1 = *(const bf16x8*)(brh + 16);
      bf16x8 l0 = *(const bf16x8*)(brl);
      bf16x8 l1 = *(const bf16x8*)(brl + 16);
      bf16x16 b_hi = __builtin_shufflevector(h0, h1, 0,1,2,3,4,5,6,7,8,9,10,11,12,13,14,15);
      bf16x16 b_lo = __builtin_shufflevector(l0, l1, 0,1,2,3,4,5,6,7,8,9,10,11,12,13,14,15);

      acc[s] = __builtin_amdgcn_wmma_f32_16x16x32_bf16(false, a_hi, false, b_hi,
                                                       (short)0, acc[s], false, false);
      acc[s] = __builtin_amdgcn_wmma_f32_16x16x32_bf16(false, a_hi, false, b_lo,
                                                       (short)0, acc[s], false, false);
      acc[s] = __builtin_amdgcn_wmma_f32_16x16x32_bf16(false, a_lo, false, b_hi,
                                                       (short)0, acc[s], false, false);
    }
    __syncthreads();
  }

  // store D: VGPR r, lanes 0-15 -> M=r, lanes 16-31 -> M=r+8; N = lane%16
  const int mbase = m0 + wave * 16 + khalf * 8;
#pragma unroll
  for (int s = 0; s < 8; ++s) {
    int col = n0 + s * 16 + ncol_lo;
#pragma unroll
    for (int r = 0; r < 8; ++r) {
      U[(size_t)(mbase + r) * N3 + col] = acc[s][r];
    }
  }
}

// -------------------- SRU scan: serial over T, parallel over B*D -----------
__global__ void __launch_bounds__(128)
sru_scan(const float* __restrict__ U,
         float* __restrict__ X,                      // in: x, out: h (in place)
         const unsigned char* __restrict__ mask,     // (B,T), 1 = valid
         const float* __restrict__ v,                // 2D
         const float* __restrict__ bias) {           // 2D
  const int gid = blockIdx.x * blockDim.x + threadIdx.x;   // 0..B*D-1
  const int b = gid >> 9;                                   // / D
  const int d = gid & (D_SZ - 1);

  const float vf = v[d];
  const float vr = v[D_SZ + d];
  const float bf = bias[d];
  const float br = bias[D_SZ + d];

  float c = 0.0f;

  const int P = 8;                       // prefetch depth (register ring)
  float pu0[P], pu1[P], pu2[P], px[P];
  unsigned char pmv[P];

#pragma unroll
  for (int j = 0; j < P; ++j) {
    size_t row = (size_t)(j * B_SZ + b) * N3;
    pu0[j] = U[row + d];
    pu1[j] = U[row + D_SZ + d];
    pu2[j] = U[row + 2 * D_SZ + d];
    px[j]  = X[(size_t)(j * B_SZ + b) * D_SZ + d];
    pmv[j] = mask[b * T_LEN + j];
  }

  for (int t0 = 0; t0 < T_LEN; t0 += P) {
#pragma unroll
    for (int j = 0; j < P; ++j) {
      const int t = t0 + j;
      const float u0 = pu0[j], u1 = pu1[j], u2 = pu2[j], xt = px[j];
      const bool valid = (pmv[j] != 0);
      const int tp = t + P;
      if (tp < T_LEN) {                   // issue loads P steps ahead
        size_t row = (size_t)(tp * B_SZ + b) * N3;
        pu0[j] = U[row + d];
        pu1[j] = U[row + D_SZ + d];
        pu2[j] = U[row + 2 * D_SZ + d];
        px[j]  = X[(size_t)(tp * B_SZ + b) * D_SZ + d];
        pmv[j] = mask[b * T_LEN + tp];
      }
      const float f = 1.0f / (1.0f + __expf(-(u1 + vf * c + bf)));
      const float r = 1.0f / (1.0f + __expf(-(u2 + vr * c + br)));
      const float c_new = f * c + (1.0f - f) * u0;
      float h = r * tanhf(c_new) + (1.0f - r) * xt;
      if (valid) c = c_new; else h = 0.0f;   // pad: keep c, zero h
      X[(size_t)(t * B_SZ + b) * D_SZ + d] = h;
    }
  }
}

// -------------------- final transpose (T,B,D) -> (B,T,D) -------------------
__global__ void sru_transpose(const float* __restrict__ X, float* __restrict__ out) {
  int linear = blockIdx.x * blockDim.x + threadIdx.x;   // over M*D/4
  int d4 = linear & 127;
  int tb = linear >> 7;
  int b  = tb & (B_SZ - 1);
  int t  = tb >> 3;
  float4 v = *(const float4*)(X + (size_t)tb * D_SZ + d4 * 4);
  *(float4*)(out + ((size_t)b * T_LEN + t) * D_SZ + d4 * 4) = v;
}

extern "C" void kernel_launch(void* const* d_in, const int* in_sizes, int n_in,
                              void* d_out, int out_size, void* d_ws, size_t ws_size,
                              hipStream_t stream) {
  (void)in_sizes; (void)n_in; (void)out_size; (void)ws_size;
  const int*           ids  = (const int*)d_in[0];
  const unsigned char* mask = (const unsigned char*)d_in[1];
  const float*         emb  = (const float*)d_in[2];
  const float*         Ws   = (const float*)d_in[3];
  const float*         vs   = (const float*)d_in[4];
  const float*         bs   = (const float*)d_in[5];
  float* out = (float*)d_out;

  float* xbuf = (float*)d_ws;                                                // 32 MB
  float* Ubuf = (float*)((char*)d_ws + (size_t)M_SZ * D_SZ * sizeof(float)); // 96 MB

  {
    int total = M_SZ * (D_SZ / 4);
    sru_gather<<<total / 256, 256, 0, stream>>>(ids, emb, xbuf);
  }
  for (int l = 0; l < L_SZ; ++l) {
    dim3 grid(N3 / BN, M_SZ / BM);
    sru_gemm<<<grid, 256, 0, stream>>>(xbuf, Ws + (size_t)l * D_SZ * N3, Ubuf);
    sru_scan<<<(B_SZ * D_SZ) / 128, 128, 0, stream>>>(
        Ubuf, xbuf, mask, vs + (size_t)l * 2 * D_SZ, bs + (size_t)l * 2 * D_SZ);
  }
  {
    int total = M_SZ * (D_SZ / 4);
    sru_transpose<<<total / 256, 256, 0, stream>>>(xbuf, out);
  }
}
